// GPT_16879221473593
// MI455X (gfx1250) — matrix operand order repaired and checked by reading, not compiled, not run on hardware
//
#include <hip/hip_runtime.h>
#include <hip/hip_bf16.h>
#include <math.h>

// ---------------- model constants ----------------
#define NLAYER 6
#define NH     12
#define DMODEL 768
#define DH     64
#define TSEQ   1024
#define BBATCH 2
#define MROWS  2048          // B*T
#define D3     2304          // 3*D
#define DFF    3072          // 4*D
#define VVOCAB 50257
#define SCALE_ATTN 0.125f    // 1/sqrt(64)
#define NEGBIG (-1.0e30f)
#define BPAD   40            // halves per LDS B row (80B: conflict-free column reads)

typedef __attribute__((ext_vector_type(16))) _Float16 v16h;
typedef __attribute__((ext_vector_type(8)))  _Float16 v8h;
typedef __attribute__((ext_vector_type(8)))  float    v8f;

__device__ __forceinline__ v8h ld8(const _Float16* p) { return *(const v8h*)p; }

__device__ __forceinline__ v16h mk16(v8h lo, v8h hi) {
  return __builtin_shufflevector(lo, hi, 0,1,2,3,4,5,6,7,8,9,10,11,12,13,14,15);
}

__device__ __forceinline__ v8f wmma_f32(v16h a, v16h b, v8f c) {
  // D = A(16x32 f16) * B(32x16 f16) + C(16x16 f32)
  return __builtin_amdgcn_wmma_f32_16x16x32_f16(false, a, false, b, (short)0, c, false, false);
}

// CDNA5 async global->LDS copy (ASYNCcnt tracked). VDST = 32-bit LDS byte
// address VGPR (flat shared pointer low 32 bits == LDS offset), VADDR = 64-bit
// global address.
__device__ __forceinline__ void async_g2l_b128(unsigned lds_addr, const void* gptr) {
  asm volatile("global_load_async_to_lds_b128 %0, %1, off"
               :: "v"(lds_addr), "v"(gptr) : "memory");
}
__device__ __forceinline__ void s_wait_async0() { asm volatile("s_wait_asynccnt 0x0" ::: "memory"); }
__device__ __forceinline__ void s_wait_async2() { asm volatile("s_wait_asynccnt 0x2" ::: "memory"); }

__device__ __forceinline__ unsigned lds_off32(const void* p) {
  return (unsigned)(unsigned long long)p;   // flat LDS addr low 32 bits = LDS byte offset
}

// ---------------- embedding: x = wte[idx] + wpe ----------------
__global__ void k_embed(const int* __restrict__ idx, const float* __restrict__ wte,
                        const float* __restrict__ wpe, float* __restrict__ x) {
  int row = blockIdx.x;            // 0..MROWS-1
  int t   = row % TSEQ;
  int tok = idx[row];
  const float* we = wte + (size_t)tok * DMODEL;
  const float* pe = wpe + (size_t)t * DMODEL;
  float* xr = x + (size_t)row * DMODEL;
  for (int d = threadIdx.x; d < DMODEL; d += blockDim.x) xr[d] = we[d] + pe[d];
}

// ---------------- layernorm -> f16 ----------------
__global__ void k_ln(const float* __restrict__ x, const float* __restrict__ w,
                     const float* __restrict__ b, _Float16* __restrict__ out) {
  __shared__ float red[256];
  int row = blockIdx.x, tid = threadIdx.x;
  const float* xr = x + (size_t)row * DMODEL;
  float s = 0.f;
  for (int d = tid; d < DMODEL; d += 256) s += xr[d];
  red[tid] = s; __syncthreads();
  for (int o = 128; o > 0; o >>= 1) { if (tid < o) red[tid] += red[tid + o]; __syncthreads(); }
  float mu = red[0] / (float)DMODEL;
  __syncthreads();
  float v = 0.f;
  for (int d = tid; d < DMODEL; d += 256) { float t = xr[d] - mu; v += t * t; }
  red[tid] = v; __syncthreads();
  for (int o = 128; o > 0; o >>= 1) { if (tid < o) red[tid] += red[tid + o]; __syncthreads(); }
  float inv = rsqrtf(red[0] / (float)DMODEL + 1e-5f);
  _Float16* orow = out + (size_t)row * DMODEL;
  for (int d = tid; d < DMODEL; d += 256)
    orow[d] = (_Float16)((xr[d] - mu) * inv * w[d] + b[d]);
}

// ---------------- weight f32[K][N] -> f16[N][K] (transpose) ----------------
__global__ void k_convT(const float* __restrict__ w, _Float16* __restrict__ o, int K, int N) {
  int i = blockIdx.x * blockDim.x + threadIdx.x;
  if (i < K * N) {
    int k = i / N, n = i % N;
    o[(size_t)n * K + k] = (_Float16)w[i];
  }
}

// ---------------- straight f32 -> f16 ----------------
__global__ void k_conv(const float* __restrict__ w, _Float16* __restrict__ o, int n) {
  int i = blockIdx.x * blockDim.x + threadIdx.x;
  if (i < n) o[i] = (_Float16)w[i];
}

// ---------------- split qkv f32[M][3D] -> q16/k16 [B,H,T,Dh], vT16 [B,H,Dh,T] ----------------
__global__ void k_split(const float* __restrict__ qkv, _Float16* __restrict__ q,
                        _Float16* __restrict__ k, _Float16* __restrict__ v) {
  int i = blockIdx.x * blockDim.x + threadIdx.x;
  if (i >= MROWS * D3) return;
  int row = i / D3, c = i % D3;
  int b = row / TSEQ, t = row % TSEQ;
  int which = c / DMODEL, cc = c % DMODEL;
  int h = cc / DH, d = cc % DH;
  int bh = b * NH + h;
  _Float16 val = (_Float16)qkv[i];
  if (which == 0)      q[((size_t)bh * TSEQ + t) * DH + d] = val;
  else if (which == 1) k[((size_t)bh * TSEQ + t) * DH + d] = val;
  else                 v[((size_t)bh * DH + d) * TSEQ + t] = val;
}

// ---------------- WMMA GEMM with async-LDS double-buffered B tiles ----------------
// C[M][N] = A[M][K](f16) x Bt[N][K](f16) + bias
// mode 0: Cf = acc+bias          (qkv)
// mode 1: Cf += acc+bias         (residual: proj, fc2; Cf = x)
// mode 2: Ch = f16(gelu(acc+b))  (fc)
// mode 3: Cf = acc (+bias)       (logits, with N tail guard)
__global__ __launch_bounds__(128) void k_gemm(const _Float16* __restrict__ A,
                                              const _Float16* __restrict__ Bt,
                                              const float* __restrict__ bias,
                                              float* __restrict__ Cf,
                                              _Float16* __restrict__ Ch,
                                              int N, int K, int mode) {
  __shared__ __attribute__((aligned(16))) _Float16 Bl[2][64][BPAD];

  int lane = threadIdx.x;
  int wv = threadIdx.y;
  int tid = wv * 32 + lane;
  int g = lane >> 4, l15 = lane & 15;
  int m0 = blockIdx.y * 64 + wv * 16;
  int nblk = blockIdx.x * 64;

  // staging role: thread covers 32B of one B row: row = tid>>1, 16-half segment (tid&1)
  int srow = tid >> 1;
  int sseg = (tid & 1) * 16;
  int nr = nblk + srow; if (nr >= N) nr = N - 1;      // clamp tail rows (logits)
  const _Float16* gB = Bt + (size_t)nr * K + sseg;
  unsigned ldsA = lds_off32(&Bl[0][srow][sseg]);
  unsigned ldsB = lds_off32(&Bl[1][srow][sseg]);

  const _Float16* arow = A + (size_t)(m0 + l15) * K;

  int nk = K >> 5;
  // prologue: stage tile 0, prefetch A regs for k=0
  async_g2l_b128(ldsA, gB);
  async_g2l_b128(ldsA + 16, gB + 8);
  v8h a_lo = ld8(arow + 8 * g);
  v8h a_hi = ld8(arow + 16 + 8 * g);

  v8f acc[4] = {};
  for (int kk = 0; kk < nk; ++kk) {
    int k0 = kk << 5;
    int cur = kk & 1;
    unsigned ldsNext = cur ? ldsA : ldsB;
    bool haveNext = (kk + 1 < nk);
    v8h an_lo = a_lo, an_hi = a_hi;
    if (haveNext) {
      // issue next B tile into the other buffer (async, ASYNCcnt)
      async_g2l_b128(ldsNext, gB + k0 + 32);
      async_g2l_b128(ldsNext + 16, gB + k0 + 40);
      // prefetch next A fragment into registers
      an_lo = ld8(arow + k0 + 32 + 8 * g);
      an_hi = ld8(arow + k0 + 48 + 8 * g);
      s_wait_async2();     // current tile's 2 loads done; next tile's 2 in flight
    } else {
      s_wait_async0();
    }
    __syncthreads();

    v16h a = mk16(a_lo, a_hi);
#pragma unroll
    for (int j = 0; j < 4; ++j) {
      const _Float16* bp = &Bl[cur][16 * j + l15][8 * g];
      v16h bb = mk16(ld8(bp), ld8(bp + 16));
      acc[j] = wmma_f32(a, bb, acc[j]);
    }
    __syncthreads();       // all waves done reading buf[cur] before it is rewritten
    a_lo = an_lo; a_hi = an_hi;
  }

#pragma unroll
  for (int j = 0; j < 4; ++j) {
    int col = nblk + 16 * j + l15;
    if (col >= N) continue;
    float bval = bias ? bias[col] : 0.f;
#pragma unroll
    for (int r = 0; r < 8; ++r) {
      int row = m0 + r + 8 * g;
      float v = acc[j][r] + bval;
      size_t o = (size_t)row * N + col;
      if (mode == 0)      Cf[o] = v;
      else if (mode == 1) Cf[o] += v;
      else if (mode == 2) Ch[o] = (_Float16)(0.5f * v * (1.f + erff(v * 0.70710678118654752f)));
      else                Cf[o] = v;
    }
  }
}

// ---------------- flash attention: one wave per (qtile, head, batch) ----------------
__global__ __launch_bounds__(32) void k_attn(const _Float16* __restrict__ q,
                                             const _Float16* __restrict__ k,
                                             const _Float16* __restrict__ vT,
                                             _Float16* __restrict__ y) {
  __shared__ __attribute__((aligned(16))) _Float16 Pl[16 * 32];
  int lane = threadIdx.x;
  int g = lane >> 4, l15 = lane & 15;
  int q0 = blockIdx.x * 16, h = blockIdx.y, b = blockIdx.z;
  int bh = b * NH + h;

  // Q tile [16 x 64] in A layout, K-chunks c=0,1
  const _Float16* qbase = q + ((size_t)bh * TSEQ + q0 + l15) * DH;
  v16h aq[2];
#pragma unroll
  for (int c = 0; c < 2; ++c)
    aq[c] = mk16(ld8(qbase + c * 32 + 8 * g), ld8(qbase + c * 32 + 16 + 8 * g));

  v8f o[4] = {};
  float mrow[8], lrow[8];
#pragma unroll
  for (int r = 0; r < 8; ++r) { mrow[r] = NEGBIG; lrow[r] = 0.f; }

  int kcMax = (q0 + 15) >> 5;
  for (int kc = 0; kc <= kcMax; ++kc) {
    int kbase = kc * 32;
    // S = Q K^T for 32 keys -> two 16x16 tiles
    v8f s[2] = {};
#pragma unroll
    for (int t = 0; t < 2; ++t) {
      int krow = kbase + 16 * t + l15;
      const _Float16* kb = k + ((size_t)bh * TSEQ + krow) * DH;
#pragma unroll
      for (int c = 0; c < 2; ++c) {
        v16h bk = mk16(ld8(kb + c * 32 + 8 * g), ld8(kb + c * 32 + 16 + 8 * g));
        s[t] = wmma_f32(aq[c], bk, s[t]);
      }
    }
    // scale + causal mask + online softmax
    float mnew[8];
#pragma unroll
    for (int r = 0; r < 8; ++r) {
      int row = q0 + r + 8 * g;
      float v0 = s[0][r] * SCALE_ATTN; if (kbase + l15 > row)      v0 = NEGBIG;
      float v1 = s[1][r] * SCALE_ATTN; if (kbase + 16 + l15 > row) v1 = NEGBIG;
      s[0][r] = v0; s[1][r] = v1;
      float mx = fmaxf(v0, v1);
#pragma unroll
      for (int off = 1; off < 16; off <<= 1) mx = fmaxf(mx, __shfl_xor(mx, off, 32));
      mnew[r] = fmaxf(mrow[r], mx);
    }
#pragma unroll
    for (int r = 0; r < 8; ++r) {
      float p0 = __expf(s[0][r] - mnew[r]);
      float p1 = __expf(s[1][r] - mnew[r]);
      float sum = p0 + p1;
#pragma unroll
      for (int off = 1; off < 16; off <<= 1) sum += __shfl_xor(sum, off, 32);
      float alpha = __expf(mrow[r] - mnew[r]);
      lrow[r] = lrow[r] * alpha + sum;
      mrow[r] = mnew[r];
#pragma unroll
      for (int j = 0; j < 4; ++j) o[j][r] *= alpha;
      int prow = r + 8 * g;
      Pl[prow * 32 + l15]      = (_Float16)p0;
      Pl[prow * 32 + 16 + l15] = (_Float16)p1;
    }
    __syncthreads();   // single-wave WG: cheap; orders LDS store->load
    // O += P V : A = P[16x32] from LDS, B = vT tiles
    v16h ap = mk16(ld8(&Pl[l15 * 32 + 16 * g]), ld8(&Pl[l15 * 32 + 16 * g + 8]));
#pragma unroll
    for (int j = 0; j < 4; ++j) {
      int d = 16 * j + l15;
      const _Float16* vb = vT + ((size_t)bh * DH + d) * TSEQ + kbase + 16 * g;
      v16h bv = mk16(ld8(vb), ld8(vb + 8));
      o[j] = wmma_f32(ap, bv, o[j]);
    }
    __syncthreads();
  }
  // normalize + store f16 to y[B*T][D]
#pragma unroll
  for (int j = 0; j < 4; ++j) {
#pragma unroll
    for (int r = 0; r < 8; ++r) {
      int row = q0 + r + 8 * g;
      float val = o[j][r] / lrow[r];
      y[((size_t)b * TSEQ + row) * DMODEL + h * DH + 16 * j + l15] = (_Float16)val;
    }
  }
}

// ---------------- host ----------------
extern "C" void kernel_launch(void* const* d_in, const int* in_sizes, int n_in,
                              void* d_out, int out_size, void* d_ws, size_t ws_size,
                              hipStream_t stream) {
  const int*   idx    = (const int*)d_in[0];
  const float* wte    = (const float*)d_in[1];
  const float* wpe    = (const float*)d_in[2];
  const float* ln1_w  = (const float*)d_in[3];
  const float* ln1_b  = (const float*)d_in[4];
  const float* attn_w = (const float*)d_in[5];
  const float* attn_b = (const float*)d_in[6];
  const float* proj_w = (const float*)d_in[7];
  const float* proj_b = (const float*)d_in[8];
  const float* ln2_w  = (const float*)d_in[9];
  const float* ln2_b  = (const float*)d_in[10];
  const float* fc_w   = (const float*)d_in[11];
  const float* fc_b   = (const float*)d_in[12];
  const float* fc2_w  = (const float*)d_in[13];
  const float* fc2_b  = (const float*)d_in[14];
  const float* lnf_w  = (const float*)d_in[15];
  const float* lnf_b  = (const float*)d_in[16];
  float* out = (float*)d_out;

  char* p = (char*)d_ws;
  auto take = [&](size_t bytes) -> void* {
    void* r = (void*)p;
    p += (bytes + 255) & ~(size_t)255;
    return r;
  };
  float*    x     = (float*)take((size_t)MROWS * DMODEL * 4);
  _Float16* a16   = (_Float16*)take((size_t)MROWS * DMODEL * 2);
  float*    qkv   = (float*)take((size_t)MROWS * D3 * 4);
  _Float16* q16   = (_Float16*)take((size_t)MROWS * DMODEL * 2);
  _Float16* k16   = (_Float16*)take((size_t)MROWS * DMODEL * 2);
  _Float16* vT16  = (_Float16*)take((size_t)MROWS * DMODEL * 2);
  _Float16* y16   = (_Float16*)take((size_t)MROWS * DMODEL * 2);
  _Float16* mlp16 = (_Float16*)take((size_t)MROWS * DFF * 2);
  _Float16* wT    = (_Float16*)take((size_t)DMODEL * DFF * 2);
  _Float16* wte16 = (_Float16*)take((size_t)VVOCAB * DMODEL * 2);

  dim3 tb(32, 4);

  // embedding + wte f16 (for logits NT GEMM)
  {
    int n = VVOCAB * DMODEL;
    k_conv<<<(n + 255) / 256, 256, 0, stream>>>(wte, wte16, n);
  }
  k_embed<<<MROWS, 256, 0, stream>>>(idx, wte, wpe, x);

  for (int l = 0; l < NLAYER; ++l) {
    // --- attention block ---
    k_ln<<<MROWS, 256, 0, stream>>>(x, ln1_w + l * DMODEL, ln1_b + l * DMODEL, a16);

    { int n = DMODEL * D3;
      k_convT<<<(n + 255) / 256, 256, 0, stream>>>(attn_w + (size_t)l * n, wT, DMODEL, D3); }
    k_gemm<<<dim3(D3 / 64, MROWS / 64), tb, 0, stream>>>(a16, wT, attn_b + l * D3,
                                                         qkv, nullptr, D3, DMODEL, 0);
    { int n = MROWS * D3;
      k_split<<<(n + 255) / 256, 256, 0, stream>>>(qkv, q16, k16, vT16); }
    k_attn<<<dim3(TSEQ / 16, NH, BBATCH), 32, 0, stream>>>(q16, k16, vT16, y16);

    { int n = DMODEL * DMODEL;
      k_convT<<<(n + 255) / 256, 256, 0, stream>>>(proj_w + (size_t)l * n, wT, DMODEL, DMODEL); }
    k_gemm<<<dim3(DMODEL / 64, MROWS / 64), tb, 0, stream>>>(y16, wT, proj_b + l * DMODEL,
                                                             x, nullptr, DMODEL, DMODEL, 1);

    // --- MLP block ---
    k_ln<<<MROWS, 256, 0, stream>>>(x, ln2_w + l * DMODEL, ln2_b + l * DMODEL, a16);

    { int n = DMODEL * DFF;
      k_convT<<<(n + 255) / 256, 256, 0, stream>>>(fc_w + (size_t)l * n, wT, DMODEL, DFF); }
    k_gemm<<<dim3(DFF / 64, MROWS / 64), tb, 0, stream>>>(a16, wT, fc_b + l * DFF,
                                                          nullptr, mlp16, DFF, DMODEL, 2);

    { int n = DFF * DMODEL;
      k_convT<<<(n + 255) / 256, 256, 0, stream>>>(fc2_w + (size_t)l * n, wT, DFF, DMODEL); }
    k_gemm<<<dim3(DMODEL / 64, MROWS / 64), tb, 0, stream>>>(mlp16, wT, fc2_b + l * DMODEL,
                                                             x, nullptr, DMODEL, DFF, 1);
  }

  // final LN + logits = lnf(x) @ wte^T
  k_ln<<<MROWS, 256, 0, stream>>>(x, lnf_w, lnf_b, a16);
  k_gemm<<<dim3((VVOCAB + 63) / 64, MROWS / 64), tb, 0, stream>>>(a16, wte16, nullptr,
                                                                  out, nullptr, VVOCAB, DMODEL, 3);
}
